// StackedLSTM_79534204387582
// MI455X (gfx1250) — matrix operand order repaired and checked by reading, not compiled, not run on hardware
//
#include <hip/hip_runtime.h>

// ---------------------------------------------------------------------------
// Stacked LSTM (4 layers, one timestep), MI455X / gfx1250, wave32 + WMMA bf16.
//
// Shapes: x[128,1024], h0/c0[4,128,1024], W_ih/W_hh[4,4096,1024], b_*[4,4096]
// out = concat( h1[3] (131072), h1s (4*131072), c1s (4*131072) )  -- f32
//
// Strategy:
//   Kernel 1 (parallel over all 4 layers):
//       P[l] = h0[l] @ W_hh[l]^T + b_ih[l] + b_hh[l]       -> d_ws (8 MB)
//   Kernel 2 (4 serial launches, layer chain):
//       gates = inp @ W_ih[l]^T + P[l] ; fused LSTM epilogue -> h1,c1
// ---------------------------------------------------------------------------

#define BATCH   128
#define HID     1024
#define G4      4096
#define KT      32          // K per WMMA step (bf16)
#define NKSTEP  (HID / KT)  // 32
#define BSTRIDE 48          // bf16 elements per staged B row (96 B: 32-aligned, bank-spread)

typedef __bf16 bf16;
typedef __attribute__((ext_vector_type(16))) __bf16 v16bf;
typedef __attribute__((ext_vector_type(8)))  __bf16 v8bf;
typedef __attribute__((ext_vector_type(8)))  float  v8f;
typedef __attribute__((ext_vector_type(4)))  float  v4f;

__device__ __forceinline__ float sigmoid_f(float x) {
  return 1.0f / (1.0f + __expf(-x));
}
__device__ __forceinline__ float tanh_f(float x) {
  x = fminf(fmaxf(x, -10.0f), 10.0f);
  float e = __expf(2.0f * x);
  return (e - 1.0f) / (e + 1.0f);
}

// A-matrix fragment (16x32 MxK, bf16), contiguous K order:
//   lanes 0-15 : M = lane,    K = k+[0,16)
//   lanes 16-31: M = lane-16, K = k+[16,32)
__device__ __forceinline__ v16bf load_a_frag(const float* __restrict__ abase,
                                             int k, int lane) {
  const int row  = lane & 15;
  const int koff = (lane >> 4) << 4;
  const float* p = abase + row * HID + k + koff;
  v4f a0 = *(const v4f*)(p);
  v4f a1 = *(const v4f*)(p + 4);
  v4f b0 = *(const v4f*)(p + 8);
  v4f b1 = *(const v4f*)(p + 12);
  v16bf r;
#pragma unroll
  for (int i = 0; i < 4; ++i) {
    r[i]      = (bf16)a0[i];
    r[i + 4]  = (bf16)a1[i];
    r[i + 8]  = (bf16)b0[i];
    r[i + 12] = (bf16)b1[i];
  }
  return r;
}

// Cooperative f32 -> bf16 staging of the B tile into LDS.
// 64 rows (4 gates x 16 hidden cols) x 32 K values. 256 threads: 8 floats each.
__device__ __forceinline__ void stage_b(bf16* __restrict__ buf,
                                        const float* __restrict__ W,
                                        int n0, int k, int tid) {
  const int rb   = tid >> 2;               // 0..63 staged row
  const int q    = tid & 3;                // K quarter
  const int gate = rb >> 4;
  const long grow = (long)(gate * HID + n0 + (rb & 15));
  const float* src = W + grow * HID + k + (q << 3);
  // Prefetch the weight row two K-steps ahead (global_prefetch_b8).
  __builtin_prefetch(src + 2 * KT, 0, 1);
  v4f s0 = *(const v4f*)(src);
  v4f s1 = *(const v4f*)(src + 4);
  v8bf d;
#pragma unroll
  for (int i = 0; i < 4; ++i) { d[i] = (bf16)s0[i]; d[i + 4] = (bf16)s1[i]; }
  *(v8bf*)(buf + rb * BSTRIDE + (q << 3)) = d;
}

// Shared GEMM core: acc[g] (g = gate 0..3) += A[16 rows] @ W[g*1024+n0 ..]^T
// A: [128,1024] f32 row-major, W: [4096,1024] f32 row-major.
__device__ __forceinline__ void gemm_acc(v8f acc[4],
                                         const float* __restrict__ A,
                                         const float* __restrict__ W,
                                         int n0, bf16* __restrict__ Bs) {
  const int tid  = threadIdx.x;
  const int lane = tid & 31;
  const float* abase = A + ((tid >> 5) << 4) * HID;  // wave-private 16 rows
  bf16* buf0 = Bs;
  bf16* buf1 = Bs + 64 * BSTRIDE;

  stage_b(buf0, W, n0, 0, tid);
  __syncthreads();

  for (int ks = 0; ks < NKSTEP; ++ks) {
    bf16* cur = (ks & 1) ? buf1 : buf0;
    bf16* nxt = (ks & 1) ? buf0 : buf1;
    if (ks + 1 < NKSTEP) stage_b(nxt, W, n0, (ks + 1) * KT, tid);

    v16bf a = load_a_frag(abase, ks * KT, lane);
#pragma unroll
    for (int g = 0; g < 4; ++g) {
      // B 32x16 layout: N = lane&15 (+16*g row group), K-half = lane>>4.
      const bf16* bp = cur + (g * 16 + (lane & 15)) * BSTRIDE + ((lane >> 4) << 4);
      v16bf b = *(const v16bf*)bp;
      acc[g] = __builtin_amdgcn_wmma_f32_16x16x32_bf16(
          /*neg_a=*/false, a, /*neg_b=*/false, b,
          /*c_mod=*/(short)0, acc[g], /*reuse_a=*/false, /*reuse_b=*/false);
    }
    __syncthreads();
  }
}

// Kernel 1: P[l] = h0[l] @ W_hh[l]^T + (b_ih[l] + b_hh[l]); all layers parallel.
__global__ void __launch_bounds__(256)
hh_gates_kernel(const float* __restrict__ h0, const float* __restrict__ Whh,
                const float* __restrict__ bih, const float* __restrict__ bhh,
                float* __restrict__ P) {
  __shared__ __align__(32) bf16 Bs[2 * 64 * BSTRIDE];
  const int  ly = blockIdx.y;
  const int  n0 = blockIdx.x * 16;
  const float* A  = h0  + (size_t)ly * BATCH * HID;
  const float* W  = Whh + (size_t)ly * G4 * HID;
  const float* bi = bih + (size_t)ly * G4;
  const float* bh = bhh + (size_t)ly * G4;
  float*       Pl = P   + (size_t)ly * BATCH * G4;

  const int tid = threadIdx.x, lane = tid & 31;
  const int m0  = (tid >> 5) << 4;

  v8f acc[4];
#pragma unroll
  for (int g = 0; g < 4; ++g) {
    const int col = g * HID + n0 + (lane & 15);
    const float bias = bi[col] + bh[col];
#pragma unroll
    for (int r = 0; r < 8; ++r) acc[g][r] = bias;
  }

  gemm_acc(acc, A, W, n0, Bs);

#pragma unroll
  for (int g = 0; g < 4; ++g) {
    const int col = g * HID + n0 + (lane & 15);
#pragma unroll
    for (int r = 0; r < 8; ++r) {
      const int m = m0 + r + ((lane >> 4) << 3);   // C/D layout: M = r + 8*(lane>=16)
      Pl[m * G4 + col] = acc[g][r];
    }
  }
}

// Kernel 2: gates = inp @ W_ih^T + P; fused LSTM epilogue -> h1, c1 (and dup).
__global__ void __launch_bounds__(256)
ih_lstm_kernel(const float* __restrict__ inp, const float* __restrict__ Wih,
               const float* __restrict__ P,   const float* __restrict__ c0,
               float* __restrict__ h1, float* __restrict__ c1,
               float* __restrict__ hdup) {
  __shared__ __align__(32) bf16 Bs[2 * 64 * BSTRIDE];
  const int n0  = blockIdx.x * 16;
  const int tid = threadIdx.x, lane = tid & 31;
  const int m0  = (tid >> 5) << 4;

  v8f acc[4];
#pragma unroll
  for (int g = 0; g < 4; ++g) {
    const int col = g * HID + n0 + (lane & 15);
#pragma unroll
    for (int r = 0; r < 8; ++r) {
      const int m = m0 + r + ((lane >> 4) << 3);
      acc[g][r] = P[m * G4 + col];
    }
  }

  gemm_acc(acc, inp, Wih, n0, Bs);

  // Fused epilogue: i/f/g/o for (m, j) live at identical (VGPR, lane) slots.
#pragma unroll
  for (int r = 0; r < 8; ++r) {
    const int m = m0 + r + ((lane >> 4) << 3);
    const int j = n0 + (lane & 15);
    const float iv = sigmoid_f(acc[0][r]);
    const float fv = sigmoid_f(acc[1][r]);
    const float gv = tanh_f(acc[2][r]);
    const float ov = sigmoid_f(acc[3][r]);
    const float cold = c0[m * HID + j];
    const float cn = fv * cold + iv * gv;
    const float hn = ov * tanh_f(cn);
    c1[m * HID + j] = cn;
    h1[m * HID + j] = hn;
    if (hdup) hdup[m * HID + j] = hn;
  }
}

extern "C" void kernel_launch(void* const* d_in, const int* in_sizes, int n_in,
                              void* d_out, int out_size, void* d_ws, size_t ws_size,
                              hipStream_t stream) {
  (void)in_sizes; (void)n_in; (void)out_size; (void)ws_size;

  const float* x   = (const float*)d_in[0];
  const float* h0  = (const float*)d_in[1];
  const float* c0  = (const float*)d_in[2];
  const float* Wih = (const float*)d_in[3];
  const float* Whh = (const float*)d_in[4];
  const float* bih = (const float*)d_in[5];
  const float* bhh = (const float*)d_in[6];

  float* out = (float*)d_out;
  const size_t PLANE = (size_t)BATCH * HID;        // 131072
  float* h1s = out + PLANE;                        // [4][128][1024]
  float* c1s = out + PLANE * 5;                    // [4][128][1024]
  float* P   = (float*)d_ws;                       // [4][128][4096] f32 = 8 MB

  // All 4 layers' recurrent GEMMs in parallel: 64 N-tiles x 4 layers = 256 WGs.
  hh_gates_kernel<<<dim3(HID / 16, 4), 256, 0, stream>>>(h0, Whh, bih, bhh, P);

  // Serial layer chain (inherent dependency): inp_{l+1} = h1_l.
  for (int l = 0; l < 4; ++l) {
    const float* inp = (l == 0) ? x : (h1s + (size_t)(l - 1) * PLANE);
    ih_lstm_kernel<<<dim3(HID / 16), 256, 0, stream>>>(
        inp,
        Wih + (size_t)l * G4 * HID,
        P   + (size_t)l * BATCH * G4,
        c0  + (size_t)l * PLANE,
        h1s + (size_t)l * PLANE,
        c1s + (size_t)l * PLANE,
        (l == 3) ? out : (float*)nullptr);
  }
}
